// RichAttentionLayer_49039936586054
// MI455X (gfx1250) — matrix operand order repaired and checked by reading, not compiled, not run on hardware
//
#include <hip/hip_runtime.h>
#include <hip/hip_bf16.h>
#include <math.h>

// ---------------------------------------------------------------------------
// DIN-style attention for MI455X (gfx1250, wave32, WMMA).
//
//   x = [k, q, k*q, k-q] (768)  ->  PReLU(x@W1+b1) -> PReLU(@W2+b2) -> @W3+b3
//   masked softmax over S, then w @ keys.
//
// Key algebraic optimization:
//   x@W1 = k@(W1a+W1d) + (k*q)@W1c + q@(W1b-W1d)
//        = k @ Wq[b]   + qvec[b]          (per-batch folded weight)
// K shrinks 768 -> 192 (4x fewer FLOPs). Kernel is HBM-bound (~629MB keys
// traffic @ 23.3TB/s ~= 27us), so we use split-bf16 (hi/lo) WMMA: 3
// v_wmma_f32_16x16x32_bf16 per product term for ~fp32 accuracy, still far
// under the memory floor.
// ---------------------------------------------------------------------------

typedef __bf16 bf16_t;
typedef bf16_t v16bf __attribute__((ext_vector_type(16)));
typedef float  v8f   __attribute__((ext_vector_type(8)));

#define BB   2048
#define SS   200
#define DD   192
#define H1N  64
#define H2N  32
#define STILES 13          // 13*16 = 208 >= 200
#define NWAVES 4
#define BLOCK  128

union FragU { unsigned int u[8]; v16bf v; };

// A-operand (16-bit 16x32): lane holds row M=lane&15; half=lane>>4.
// VGPR p (packed K,K+1):  K = kbase + (p<4?0:16) + 2*(p&3),  kbase = kk*32 + half*8
__device__ __forceinline__ v16bf load_fragA(const bf16_t* __restrict__ row, int kbase) {
  FragU f;
#pragma unroll
  for (int p = 0; p < 8; ++p) {
    int K = kbase + ((p & 4) << 2) + ((p & 3) << 1);
    f.u[p] = *reinterpret_cast<const unsigned int*>(row + K);
  }
  return f.v;
}

// Contiguous-K loader (unused; B uses load_fragA): col N=lane&15; lanes 0-15: K=0..15,
// lanes 16-31: K=16..31 (not AMD's table order).  K = kbase + 2*p,
// kbase = kk*32 + half*16.  Row pointer is the [N][K] row (K contiguous).
__device__ __forceinline__ v16bf load_fragB(const bf16_t* __restrict__ row, int kbase) {
  FragU f;
#pragma unroll
  for (int p = 0; p < 8; ++p)
    f.u[p] = *reinterpret_cast<const unsigned int*>(row + kbase + (p << 1));
  return f.v;
}

#define WMMA_BF16(A, B, C) \
  __builtin_amdgcn_wmma_f32_16x16x32_bf16(false, (A), false, (B), (short)0, (C), false, false)

__device__ __forceinline__ void split_bf16(float v, bf16_t& hi, bf16_t& lo) {
  hi = (bf16_t)v;
  lo = (bf16_t)(v - (float)hi);
}

__global__ __launch_bounds__(BLOCK) void din_attn_kernel(
    const float* __restrict__ query, const float* __restrict__ keys,
    const int*   __restrict__ kmask, const float* __restrict__ W1,
    const float* __restrict__ b1,    const float* __restrict__ a1p,
    const float* __restrict__ W2,    const float* __restrict__ b2,
    const float* __restrict__ a2p,   const float* __restrict__ W3,
    const float* __restrict__ b3p,
    float* __restrict__ outWS,       // [B, D]
    float* __restrict__ outW)        // [B, S]
{
  // ---- LDS (≈123 KB; 320 KB/WGP -> 2 blocks per WGP) ----
  __shared__ __align__(16) bf16_t sWqHi[H1N * DD];   // [h][d]
  __shared__ __align__(16) bf16_t sWqLo[H1N * DD];
  __shared__ __align__(16) bf16_t sW2Hi[H2N * H1N];  // [n][k]
  __shared__ __align__(16) bf16_t sW2Lo[H2N * H1N];
  __shared__ __align__(16) bf16_t sAHi[NWAVES * 16 * DD];   // per-wave key tile [m][d]
  __shared__ __align__(16) bf16_t sALo[NWAVES * 16 * DD];
  __shared__ __align__(16) bf16_t sH1Hi[NWAVES * 16 * H1N]; // per-wave h1 tile [m][k]
  __shared__ __align__(16) bf16_t sH1Lo[NWAVES * 16 * H1N];
  __shared__ float sQ[DD];
  __shared__ float sQvec[H1N];
  __shared__ float sW3[H2N];
  __shared__ float sB2[H2N];
  __shared__ float sScores[STILES * 16];
  __shared__ float sWgt[STILES * 16];
  __shared__ float sRed[BLOCK];

  const int b    = blockIdx.x;
  const int tid  = threadIdx.x;
  const int lane = tid & 31;
  const int wave = tid >> 5;
  const float NEG_INF = -__builtin_inff();

  const float a1v = a1p[0];
  const float a2v = a2p[0];
  const float b3v = b3p[0];

  // ---- setup: q into LDS ----
  for (int i = tid; i < DD; i += BLOCK) sQ[i] = query[(size_t)b * DD + i];
  __syncthreads();

  // Wq[h][d] = W1a + W1d + q[d]*W1c   (split hi/lo bf16)
  for (int i = tid; i < DD * H1N; i += BLOCK) {
    int d = i >> 6, h = i & 63;
    float w = W1[(size_t)d * H1N + h] + W1[(size_t)(576 + d) * H1N + h] +
              sQ[d] * W1[(size_t)(384 + d) * H1N + h];
    split_bf16(w, sWqHi[h * DD + d], sWqLo[h * DD + d]);
  }
  // W2 transposed [n][k]
  for (int i = tid; i < H1N * H2N; i += BLOCK) {
    int k = i >> 5, n = i & 31;
    split_bf16(W2[i], sW2Hi[n * H1N + k], sW2Lo[n * H1N + k]);
  }
  if (tid < H2N) { sW3[tid] = W3[tid]; sB2[tid] = b2[tid]; }
  // qvec[h] = b1[h] + sum_d q[d]*(W1b - W1d)
  if (tid < H1N) {
    float acc = b1[tid];
    for (int d = 0; d < DD; ++d)
      acc += sQ[d] * (W1[(size_t)(192 + d) * H1N + tid] -
                      W1[(size_t)(576 + d) * H1N + tid]);
    sQvec[tid] = acc;
  }
  __syncthreads();

  // ---- per-wave S-tiles: MLP via split-bf16 WMMA ----
  const int half = lane >> 4;
  const int mrow = lane & 15;

  for (int t = wave; t < STILES; t += NWAVES) {
    const int s0 = t * 16;
    bf16_t* aHiT = sAHi + wave * (16 * DD);
    bf16_t* aLoT = sALo + wave * (16 * DD);

    // stage keys tile (coalesced), convert + split
    for (int idx = lane; idx < 16 * DD; idx += 32) {
      int r = idx / DD, d = idx - r * DD;
      int s = s0 + r;
      float v = (s < SS) ? keys[((size_t)b * SS + s) * DD + d] : 0.0f;
      split_bf16(v, aHiT[idx], aLoT[idx]);
    }
    asm volatile("s_wait_dscnt 0" ::: "memory");  // same-wave LDS RAW

    // layer 1: 16x64 tile, K=192: 4 coltiles x 6 ksteps x 3 split WMMAs
    v8f c[4];
#pragma unroll
    for (int ct = 0; ct < 4; ++ct)
#pragma unroll
      for (int j = 0; j < 8; ++j) c[ct][j] = 0.0f;

    const bf16_t* aHiRow = aHiT + mrow * DD;
    const bf16_t* aLoRow = aLoT + mrow * DD;
#pragma unroll
    for (int kk = 0; kk < 6; ++kk) {
      v16bf aH = load_fragA(aHiRow, kk * 32 + half * 8);
      v16bf aL = load_fragA(aLoRow, kk * 32 + half * 8);
#pragma unroll
      for (int ct = 0; ct < 4; ++ct) {
        const bf16_t* bHiRow = sWqHi + (ct * 16 + mrow) * DD;
        const bf16_t* bLoRow = sWqLo + (ct * 16 + mrow) * DD;
        v16bf bH = load_fragA(bHiRow, kk * 32 + half * 8);
        v16bf bL = load_fragA(bLoRow, kk * 32 + half * 8);
        c[ct] = WMMA_BF16(aH, bH, c[ct]);
        c[ct] = WMMA_BF16(aH, bL, c[ct]);
        c[ct] = WMMA_BF16(aL, bH, c[ct]);
      }
    }

    // bias + PReLU(a1), stage h1 (C/D layout: M = r + 8*half, N = lane&15)
    bf16_t* h1HiT = sH1Hi + wave * (16 * H1N);
    bf16_t* h1LoT = sH1Lo + wave * (16 * H1N);
#pragma unroll
    for (int ct = 0; ct < 4; ++ct) {
      int hcol = ct * 16 + mrow;
      float qv = sQvec[hcol];
#pragma unroll
      for (int r = 0; r < 8; ++r) {
        int M = r + 8 * half;
        float v = c[ct][r] + qv;
        v = (v >= 0.0f) ? v : a1v * v;
        split_bf16(v, h1HiT[M * H1N + hcol], h1LoT[M * H1N + hcol]);
      }
    }
    asm volatile("s_wait_dscnt 0" ::: "memory");

    // layer 2: 16x32, K=64: 2 coltiles x 2 ksteps x 3 split WMMAs
    v8f c2[2];
#pragma unroll
    for (int ct = 0; ct < 2; ++ct)
#pragma unroll
      for (int j = 0; j < 8; ++j) c2[ct][j] = 0.0f;

    const bf16_t* a2HiRow = h1HiT + mrow * H1N;
    const bf16_t* a2LoRow = h1LoT + mrow * H1N;
#pragma unroll
    for (int kk = 0; kk < 2; ++kk) {
      v16bf aH = load_fragA(a2HiRow, kk * 32 + half * 8);
      v16bf aL = load_fragA(a2LoRow, kk * 32 + half * 8);
#pragma unroll
      for (int ct = 0; ct < 2; ++ct) {
        const bf16_t* bHiRow = sW2Hi + (ct * 16 + mrow) * H1N;
        const bf16_t* bLoRow = sW2Lo + (ct * 16 + mrow) * H1N;
        v16bf bH = load_fragA(bHiRow, kk * 32 + half * 8);
        v16bf bL = load_fragA(bLoRow, kk * 32 + half * 8);
        c2[ct] = WMMA_BF16(aH, bH, c2[ct]);
        c2[ct] = WMMA_BF16(aH, bL, c2[ct]);
        c2[ct] = WMMA_BF16(aL, bH, c2[ct]);
      }
    }

    // layer 3: bias + PReLU(a2), dot with W3, shfl-reduce across 16 lanes
    float w3c0 = sW3[mrow], w3c1 = sW3[16 + mrow];
    float bb0  = sB2[mrow], bb1  = sB2[16 + mrow];
#pragma unroll
    for (int r = 0; r < 8; ++r) {
      float v0 = c2[0][r] + bb0; v0 = (v0 >= 0.0f) ? v0 : a2v * v0;
      float v1 = c2[1][r] + bb1; v1 = (v1 >= 0.0f) ? v1 : a2v * v1;
      float p  = v0 * w3c0 + v1 * w3c1;
      p += __shfl_xor(p, 8, 16);
      p += __shfl_xor(p, 4, 16);
      p += __shfl_xor(p, 2, 16);
      p += __shfl_xor(p, 1, 16);
      if (mrow == 0) sScores[s0 + r + 8 * half] = p + b3v;
    }
    asm volatile("s_wait_dscnt 0" ::: "memory");
  }
  __syncthreads();

  // ---- masked softmax over S (all-masked row -> w = 0, matching reference)
  float lmax = NEG_INF;
  for (int s = tid; s < STILES * 16; s += BLOCK) {
    float sc = NEG_INF;
    if (s < SS && kmask[(size_t)b * SS + s] != 0) sc = sScores[s];
    sScores[s] = sc;
    lmax = fmaxf(lmax, sc);
  }
  sRed[tid] = lmax;
  __syncthreads();
  for (int off = BLOCK / 2; off > 0; off >>= 1) {
    if (tid < off) sRed[tid] = fmaxf(sRed[tid], sRed[tid + off]);
    __syncthreads();
  }
  float gmax = sRed[0];
  __syncthreads();

  float lsum = 0.0f;
  for (int s = tid; s < SS; s += BLOCK) {
    float sc = sScores[s];
    float e  = 0.0f;
    if (gmax != NEG_INF && sc != NEG_INF) e = expf(sc - gmax);
    sWgt[s] = e;
    lsum += e;
  }
  sRed[tid] = lsum;
  __syncthreads();
  for (int off = BLOCK / 2; off > 0; off >>= 1) {
    if (tid < off) sRed[tid] += sRed[tid + off];
    __syncthreads();
  }
  float gsum = sRed[0];
  __syncthreads();

  float inv = (gsum > 0.0f) ? (1.0f / gsum) : 0.0f;
  for (int s = tid; s < SS; s += BLOCK) {
    float w = sWgt[s] * inv;
    sWgt[s] = w;
    outW[(size_t)b * SS + s] = w;
  }
  __syncthreads();

  // ---- weighted sum: out[b,d] = sum_s w[s]*keys[b,s,d] (coalesced over d)
  for (int d = tid; d < DD; d += BLOCK) {
    float acc = 0.0f;
    const float* kb = keys + (size_t)b * SS * DD + d;
    for (int s = 0; s < SS; ++s) acc += sWgt[s] * kb[(size_t)s * DD];
    outWS[(size_t)b * DD + d] = acc;
  }
}

extern "C" void kernel_launch(void* const* d_in, const int* in_sizes, int n_in,
                              void* d_out, int out_size, void* d_ws, size_t ws_size,
                              hipStream_t stream) {
  const float* query = (const float*)d_in[0];
  const float* keys  = (const float*)d_in[1];
  const int*   kmask = (const int*)d_in[2];
  const float* W1 = (const float*)d_in[3];
  const float* b1 = (const float*)d_in[4];
  const float* a1 = (const float*)d_in[5];
  const float* W2 = (const float*)d_in[6];
  const float* b2 = (const float*)d_in[7];
  const float* a2 = (const float*)d_in[8];
  const float* W3 = (const float*)d_in[9];
  const float* b3 = (const float*)d_in[10];
  float* out   = (float*)d_out;
  float* outWS = out;                           // weighted_sum: B*D floats
  float* outW  = out + (size_t)BB * DD;         // w: B*S floats
  din_attn_kernel<<<dim3(BB), dim3(BLOCK), 0, stream>>>(
      query, keys, kmask, W1, b1, a1, W2, b2, a2, W3, b3, outWS, outW);
}
